// Gudi_UpProj_Block_13580686590343
// MI455X (gfx1250) — matrix operand order repaired and checked by reading, not compiled, not run on hardware
//
#include <hip/hip_runtime.h>

typedef __attribute__((ext_vector_type(16))) __bf16 v16bf;
typedef __attribute__((ext_vector_type(8)))  float  v8f;

union V16BF { v16bf v; uint4 q[2]; };

#define WC_HALVES (25*8*16*32*16)   /* 1,638,400 */
#define W2_HALVES (9*4*8*32*16)     /*   147,456 */

// ---------------------------------------------------------------------------
// Weight prep: fragment-ordered bf16 WMMA-B tiles.
// Tile t = (tap*NCC + cc)*NT + ntile ; layout [t][lane(32)][j(16)],
// element (K = (lane>>4)*16 + j, N = ntile*16 + (lane&15)), K = channel-in-chunk.
// ---------------------------------------------------------------------------
__global__ __launch_bounds__(256) void prep_wc_kernel(
        const float* __restrict__ w1, const float* __restrict__ wsc,
        __bf16* __restrict__ Wc) {
    int idx = blockIdx.x * 256 + threadIdx.x;
    if (idx >= WC_HALVES) return;
    int j    = idx & 15;
    int lane = (idx >> 4) & 31;
    int nt   = (idx >> 9) & 15;
    int cc   = (idx >> 13) & 7;
    int tap  = idx >> 16;                 // 0..24 = kh*5+kw
    int K = (lane >> 4) * 16 + j;
    int c = cc * 32 + K;                  // input channel 0..255
    int n = nt * 16 + (lane & 15);        // 0..127 -> w1, 128..255 -> wsc
    int kh = tap / 5, kw = tap % 5;
    float v = (n < 128)
        ? w1 [(((size_t)n       * 256 + c) * 5 + kh) * 5 + kw]
        : wsc[(((size_t)(n-128) * 256 + c) * 5 + kh) * 5 + kw];
    Wc[idx] = (__bf16)v;
}

__global__ __launch_bounds__(256) void prep_w2_kernel(
        const float* __restrict__ w2, __bf16* __restrict__ W2) {
    int idx = blockIdx.x * 256 + threadIdx.x;
    if (idx >= W2_HALVES) return;
    int j    = idx & 15;
    int lane = (idx >> 4) & 31;
    int nt   = (idx >> 9) & 7;
    int cc   = (idx >> 12) & 3;
    int tap  = idx >> 14;                 // 0..8 = kh*3+kw
    int K = (lane >> 4) * 16 + j;
    int c = cc * 32 + K;
    int n = nt * 16 + (lane & 15);
    int kh = tap / 3, kw = tap % 3;
    W2[idx] = (__bf16)w2[(((size_t)n * 128 + c) * 3 + kh) * 3 + kw];
}

// ---------------------------------------------------------------------------
// Conv1 + shortcut conv (5x5 over 2x-dilated input), parity-decomposed.
// Output at (2i+ph, 2j+pw) = small conv over original 32x32 x.
// Block: one (image, parity, 2-row strip). M=64 pixels, N=256 (y1||ysc).
// 8 waves: wave = (wm<1..0>, ns<1:0>): 2 M-tiles x 4 N-tiles each.
// ---------------------------------------------------------------------------
__global__ __launch_bounds__(256) void conv1_kernel(
        const float* __restrict__ x, const __bf16* __restrict__ Wc,
        float* __restrict__ y1, float* __restrict__ ysc) {
    const int rb = blockIdx.x;            // 0..15 (row pair)
    const int p  = blockIdx.y;            // 0..3 parity
    const int b  = blockIdx.z;            // image
    const int ph = p >> 1, pw = p & 1;
    const int i0 = rb * 2;
    const int tid  = threadIdx.x;
    const int lane = tid & 31;
    const int wv   = tid >> 5;
    const int wm   = wv >> 2;             // 0..1
    const int ns   = wv & 3;              // 0..3
    const int half = lane >> 4;
    const int m    = lane & 15;

    // x tile: rows i0-1..i0+2, cols -1..32, 32 channels; pixel-major bf16
    __shared__ __align__(32) __bf16 xs[4 * 34 * 32];

    int dis[3], khs[3], djs[3], kws[3], ndi, ndj;
    if (ph) { ndi = 2; dis[0] = 0; khs[0] = 1; dis[1] = 1; khs[1] = 3;
              dis[2] = 0; khs[2] = 0; }
    else    { ndi = 3; dis[0] = -1; khs[0] = 0; dis[1] = 0; khs[1] = 2;
              dis[2] = 1; khs[2] = 4; }
    if (pw) { ndj = 2; djs[0] = 0; kws[0] = 1; djs[1] = 1; kws[1] = 3;
              djs[2] = 0; kws[2] = 0; }
    else    { ndj = 3; djs[0] = -1; kws[0] = 0; djs[1] = 0; kws[1] = 2;
              djs[2] = 1; kws[2] = 4; }

    v8f acc[2][4];
    const v8f vzero = {0.f,0.f,0.f,0.f,0.f,0.f,0.f,0.f};
    #pragma unroll
    for (int a = 0; a < 2; ++a)
        #pragma unroll
        for (int c2 = 0; c2 < 4; ++c2) acc[a][c2] = vzero;

    for (int cc = 0; cc < 8; ++cc) {       // 8 chunks of 32 input channels
        __syncthreads();
        for (int idx = tid; idx < 4 * 34 * 32; idx += 256) {
            int c   = idx / 136;           // 136 = 4*34 pixels
            int pix = idx - c * 136;
            int r   = pix / 34;
            int cp  = pix - r * 34;
            int gr  = i0 - 1 + r;
            int gc  = cp - 1;
            float v = 0.0f;
            if ((unsigned)gr < 32u && (unsigned)gc < 32u)
                v = x[(((size_t)b * 256 + (cc * 32 + c)) * 32 + gr) * 32 + gc];
            xs[pix * 32 + c] = (__bf16)v;
        }
        __syncthreads();

        for (int ti = 0; ti < ndi; ++ti) {
            for (int tj = 0; tj < ndj; ++tj) {
                int tap = khs[ti] * 5 + kws[tj];
                const __bf16* bbase = Wc + (size_t)((tap * 8 + cc) * 16) * 512;
                __builtin_prefetch(bbase + (size_t)16 * 512, 0, 1);  // next chunk's tiles
                v16bf bw[4];
                #pragma unroll
                for (int nt = 0; nt < 4; ++nt)
                    bw[nt] = *reinterpret_cast<const v16bf*>(
                        bbase + (size_t)(ns * 4 + nt) * 512 + lane * 16);
                #pragma unroll
                for (int mt = 0; mt < 2; ++mt) {
                    int mg   = (wm * 2 + mt) * 16 + m;
                    int mrow = mg >> 5;
                    int jj   = mg & 31;
                    int pix  = (mrow + dis[ti] + 1) * 34 + (jj + djs[tj] + 1);
                    const __bf16* ap = xs + pix * 32 + half * 16;
                    V16BF af;
                    af.q[0] = *reinterpret_cast<const uint4*>(ap);        // K=half*16..+7
                    af.q[1] = *reinterpret_cast<const uint4*>(ap + 8);    // K=half*16+8..+15
                    #pragma unroll
                    for (int nt = 0; nt < 4; ++nt)
                        acc[mt][nt] = __builtin_amdgcn_wmma_f32_16x16x32_bf16(
                            false, af.v, false, bw[nt], (short)0,
                            acc[mt][nt], false, false);
                }
            }
        }
    }

    // D layout: N = lane&15, M = r + 8*(lane>>4)
    #pragma unroll
    for (int mt = 0; mt < 2; ++mt)
        #pragma unroll
        for (int nt = 0; nt < 4; ++nt) {
            int nglob = (ns * 4 + nt) * 16 + m;
            float* dst = (nglob < 128) ? y1 : ysc;
            int n = nglob & 127;
            #pragma unroll
            for (int r = 0; r < 8; ++r) {
                int mg   = (wm * 2 + mt) * 16 + r + 8 * half;
                int mrow = mg >> 5;
                int jj   = mg & 31;
                int oh   = 2 * (i0 + mrow) + ph;
                int ow   = 2 * jj + pw;
                dst[(((size_t)b * 128 + n) * 64 + oh) * 64 + ow] = acc[mt][nt][r];
            }
        }
}

// ---------------------------------------------------------------------------
// Conv2: dense 3x3 on relu(bn1(y1)) applied during LDS staging.
// Block: (image, 2-row strip of 64 cols) -> M=128 pixels, N=128.
// 8 waves: wm=wv>>1 (0..3), ns=wv&1; 2 M-tiles x 4 N-tiles each.
// ---------------------------------------------------------------------------
__global__ __launch_bounds__(256) void conv2_kernel(
        const float* __restrict__ y1, const __bf16* __restrict__ W2,
        const float* __restrict__ scale1, const float* __restrict__ shift1,
        float* __restrict__ y2) {
    const int rb = blockIdx.x;            // 0..31
    const int b  = blockIdx.y;
    const int oh0 = rb * 2;
    const int tid  = threadIdx.x;
    const int lane = tid & 31;
    const int wv   = tid >> 5;
    const int wm   = wv >> 1;             // 0..3
    const int ns   = wv & 1;              // 0..1
    const int half = lane >> 4;
    const int m    = lane & 15;

    __shared__ __align__(32) __bf16 xs[4 * 66 * 32];   // rows oh0-1..+2, cols -1..64

    v8f acc[2][4];
    const v8f vzero = {0.f,0.f,0.f,0.f,0.f,0.f,0.f,0.f};
    #pragma unroll
    for (int a = 0; a < 2; ++a)
        #pragma unroll
        for (int c2 = 0; c2 < 4; ++c2) acc[a][c2] = vzero;

    for (int cc = 0; cc < 4; ++cc) {       // 4 chunks of 32 channels
        __syncthreads();
        for (int idx = tid; idx < 4 * 66 * 32; idx += 256) {
            int c   = idx / 264;           // 264 = 4*66 pixels
            int pix = idx - c * 264;
            int r   = pix / 66;
            int cp  = pix - r * 66;
            int gr  = oh0 - 1 + r;
            int gc  = cp - 1;
            int cg  = cc * 32 + c;
            float v = 0.0f;
            if ((unsigned)gr < 64u && (unsigned)gc < 64u) {
                v = y1[(((size_t)b * 128 + cg) * 64 + gr) * 64 + gc];
                v = fmaxf(v * scale1[cg] + shift1[cg], 0.0f);   // BN1 + ReLU
            }
            xs[pix * 32 + c] = (__bf16)v;
        }
        __syncthreads();

        for (int tap = 0; tap < 9; ++tap) {
            int di = tap / 3 - 1, dj = tap % 3 - 1;
            const __bf16* bbase = W2 + (size_t)((tap * 4 + cc) * 8) * 512;
            __builtin_prefetch(bbase + (size_t)8 * 512, 0, 1);
            v16bf bw[4];
            #pragma unroll
            for (int nt = 0; nt < 4; ++nt)
                bw[nt] = *reinterpret_cast<const v16bf*>(
                    bbase + (size_t)(ns * 4 + nt) * 512 + lane * 16);
            #pragma unroll
            for (int mt = 0; mt < 2; ++mt) {
                int mg   = (wm * 2 + mt) * 16 + m;
                int mrow = mg >> 6;
                int col  = mg & 63;
                int pix  = (mrow + di + 1) * 66 + (col + dj + 1);
                const __bf16* ap = xs + pix * 32 + half * 16;
                V16BF af;
                af.q[0] = *reinterpret_cast<const uint4*>(ap);
                af.q[1] = *reinterpret_cast<const uint4*>(ap + 8);
                #pragma unroll
                for (int nt = 0; nt < 4; ++nt)
                    acc[mt][nt] = __builtin_amdgcn_wmma_f32_16x16x32_bf16(
                        false, af.v, false, bw[nt], (short)0,
                        acc[mt][nt], false, false);
            }
        }
    }

    #pragma unroll
    for (int mt = 0; mt < 2; ++mt)
        #pragma unroll
        for (int nt = 0; nt < 4; ++nt) {
            int n = (ns * 4 + nt) * 16 + m;
            #pragma unroll
            for (int r = 0; r < 8; ++r) {
                int mg   = (wm * 2 + mt) * 16 + r + 8 * half;
                int mrow = mg >> 6;
                int col  = mg & 63;
                y2[(((size_t)b * 128 + n) * 64 + (oh0 + mrow)) * 64 + col] =
                    acc[mt][nt][r];
            }
        }
}

// ---------------------------------------------------------------------------
// Deterministic per-channel sum / sumsq (one block per channel).
// ---------------------------------------------------------------------------
__global__ __launch_bounds__(256) void reduce_stats_kernel(
        const float* __restrict__ y, float* __restrict__ sum,
        float* __restrict__ sq, int nch) {
    int ch  = blockIdx.x;
    int tid = threadIdx.x;
    float s = 0.f, q = 0.f;
    for (int b = 0; b < 8; ++b) {
        const float* p = y + ((size_t)b * nch + ch) * 4096;
        for (int i = tid; i < 4096; i += 256) { float v = p[i]; s += v; q += v * v; }
    }
    __shared__ float ss[256], qq[256];
    ss[tid] = s; qq[tid] = q;
    __syncthreads();
    for (int off = 128; off > 0; off >>= 1) {
        if (tid < off) { ss[tid] += ss[tid + off]; qq[tid] += qq[tid + off]; }
        __syncthreads();
    }
    if (tid == 0) { sum[ch] = ss[0]; sq[ch] = qq[0]; }
}

__global__ void finalize_bn_kernel(
        const float* __restrict__ sum, const float* __restrict__ sq,
        const float* __restrict__ gamma, const float* __restrict__ beta,
        float* __restrict__ scale, float* __restrict__ shift, float inv_count) {
    int c = threadIdx.x;
    float mean = sum[c] * inv_count;
    float var  = sq[c] * inv_count - mean * mean;
    float is   = rsqrtf(var + 1e-5f);
    float sc   = gamma[c] * is;
    scale[c] = sc;
    shift[c] = beta[c] - mean * sc;
}

// out = relu(bn2(y2) + bn_sc(ysc)), in place over d_out (holds raw y2)
__global__ __launch_bounds__(256) void final_kernel(
        float* __restrict__ out, const float* __restrict__ ysc,
        const float* __restrict__ s2, const float* __restrict__ h2,
        const float* __restrict__ ss, const float* __restrict__ hs) {
    const size_t total = (size_t)8 * 128 * 64 * 64;
    for (size_t e = (size_t)blockIdx.x * 256 + threadIdx.x; e < total;
         e += (size_t)gridDim.x * 256) {
        int n = (int)((e >> 12) & 127);
        float v = out[e] * s2[n] + h2[n] + ysc[e] * ss[n] + hs[n];
        out[e] = fmaxf(v, 0.0f);
    }
}

// ---------------------------------------------------------------------------
extern "C" void kernel_launch(void* const* d_in, const int* in_sizes, int n_in,
                              void* d_out, int out_size, void* d_ws, size_t ws_size,
                              hipStream_t stream) {
    (void)in_sizes; (void)n_in; (void)out_size; (void)ws_size;
    const float* x   = (const float*)d_in[0];
    const float* w1  = (const float*)d_in[1];
    const float* g1  = (const float*)d_in[2];
    const float* b1  = (const float*)d_in[3];
    const float* w2  = (const float*)d_in[4];
    const float* g2  = (const float*)d_in[5];
    const float* b2  = (const float*)d_in[6];
    const float* wsc = (const float*)d_in[7];
    const float* gsc = (const float*)d_in[8];
    const float* bsc = (const float*)d_in[9];
    float* out = (float*)d_out;

    const size_t NOUT = (size_t)8 * 128 * 64 * 64;   // 4,194,304
    char* ws = (char*)d_ws;                          // needs ~37.2 MB
    float*  y1   = (float*)ws;   ws += NOUT * 4;
    float*  ysc  = (float*)ws;   ws += NOUT * 4;
    __bf16* Wc   = (__bf16*)ws;  ws += (size_t)WC_HALVES * 2;   // 32B-aligned
    __bf16* W2   = (__bf16*)ws;  ws += (size_t)W2_HALVES * 2;
    float* sumA  = (float*)ws;   ws += 128 * 4;
    float* sqA   = (float*)ws;   ws += 128 * 4;
    float* sumS  = (float*)ws;   ws += 128 * 4;
    float* sqS   = (float*)ws;   ws += 128 * 4;
    float* sum2  = (float*)ws;   ws += 128 * 4;
    float* sq2   = (float*)ws;   ws += 128 * 4;
    float* scale1  = (float*)ws; ws += 128 * 4;
    float* shift1  = (float*)ws; ws += 128 * 4;
    float* scalesc = (float*)ws; ws += 128 * 4;
    float* shiftsc = (float*)ws; ws += 128 * 4;
    float* scale2  = (float*)ws; ws += 128 * 4;
    float* shift2  = (float*)ws; ws += 128 * 4;

    const float invc = 1.0f / 32768.0f;   // B*OH*OW

    prep_wc_kernel<<<WC_HALVES / 256, 256, 0, stream>>>(w1, wsc, Wc);
    prep_w2_kernel<<<W2_HALVES / 256, 256, 0, stream>>>(w2, W2);

    conv1_kernel<<<dim3(16, 4, 8), 256, 0, stream>>>(x, Wc, y1, ysc);

    reduce_stats_kernel<<<128, 256, 0, stream>>>(y1,  sumA, sqA, 128);
    reduce_stats_kernel<<<128, 256, 0, stream>>>(ysc, sumS, sqS, 128);
    finalize_bn_kernel<<<1, 128, 0, stream>>>(sumA, sqA, g1,  b1,  scale1,  shift1,  invc);
    finalize_bn_kernel<<<1, 128, 0, stream>>>(sumS, sqS, gsc, bsc, scalesc, shiftsc, invc);

    conv2_kernel<<<dim3(32, 8), 256, 0, stream>>>(y1, W2, scale1, shift1, out);

    reduce_stats_kernel<<<128, 256, 0, stream>>>(out, sum2, sq2, 128);
    finalize_bn_kernel<<<1, 128, 0, stream>>>(sum2, sq2, g2, b2, scale2, shift2, invc);

    final_kernel<<<2048, 256, 0, stream>>>(out, ysc, scale2, shift2, scalesc, shiftsc);
}
